// Attention_20882130993856
// MI455X (gfx1250) — matrix operand order repaired and checked by reading, not compiled, not run on hardware
//
#include <hip/hip_runtime.h>

// FlashAttention forward, causal, GQA (H=16, KH=4), D=128, fp32 in/out,
// f16 WMMA compute on gfx1250 (wave32).
//
// Workgroup = one GQA group: 4 waves = the 4 query heads sharing one KV head,
// all on the same 32-query tile -> identical causal trip counts (uniform
// barriers) and K/V tiles are loaded+converted+transposed into LDS ONCE per
// workgroup, amortized over 4 heads x 2 query sets = 8 WMMAs per fragment.
//
//   S^T(keys x q) = K_tile(16k x D) * Q^T(D x 16q)   -> 16x v_wmma per tile/wave
//   O^T(d x q)   += V^T(16d x 32k)  * P^T(32k x 16q) -> 16x v_wmma per tile/wave

typedef __attribute__((ext_vector_type(16))) _Float16 v16h;
typedef __attribute__((ext_vector_type(8)))  float    v8f;

#define NH      16
#define NKH     4
#define HPK     (NH / NKH)    // query heads per KV head
#define HD      128
#define KVSTR   (NKH * HD)    // floats per token in k / v
#define QSTR    (NH * HD)     // floats per token in q / out
#define NEG_BIG (-1.0e30f)
// softmax scale (1/sqrt(128)) folded with log2(e) so exp is raw v_exp_f32 (2^x)
#define QK_SCALE_LOG2 (0.08838834764831845f * 1.4426950408889634f)

// LDS strides in f16 halves; both keep 16B alignment for b128 access and give
// bank-conflict-free fragment reads (68-dword and 20-dword lane strides).
#define KSTR 136    // K tile row stride:  [key][d],  32 x 136 halves = 8704 B
#define VSTR 40     // V^T tile row stride:[d][key], 128 x  40 halves = 10240 B

union H16 { v16h v; _Float16 h[16]; unsigned u[8]; };
union F8  { v8f  v; float    f[8]; };

__device__ __forceinline__ float4 ld_f4(const float* p) { return *(const float4*)p; }

__device__ __forceinline__ unsigned pack_h2(float a, float b) {
  union { _Float16 h[2]; unsigned u; } t;
  t.h[0] = (_Float16)a; t.h[1] = (_Float16)b;
  return t.u;
}

// Read a 16x32 f16 A-fragment slice from LDS: halves 0..7 at base, 8..15 at
// base+16 halves (two ds_load_b128).
__device__ __forceinline__ void lds_frag(H16& f, const _Float16* base) {
  uint4 a = *(const uint4*)base;
  uint4 b = *(const uint4*)(base + 16);
  f.u[0] = a.x; f.u[1] = a.y; f.u[2] = a.z; f.u[3] = a.w;
  f.u[4] = b.x; f.u[5] = b.y; f.u[6] = b.z; f.u[7] = b.w;
}

__global__ __launch_bounds__(128, 1)
void fa_fwd_causal_gqa(const float* __restrict__ q, const float* __restrict__ k,
                       const float* __restrict__ v, const int* __restrict__ cu,
                       float* __restrict__ out, int qtiles2)
{
  __shared__ _Float16 ldsK[32 * KSTR];   // f16 K tile, [key][d]
  __shared__ _Float16 ldsV[HD * VSTR];   // f16 V^T tile, [d][key]

  const int tid  = threadIdx.x;
  const int lane = tid & 31;
  const int n    = lane & 15;       // query column owned by this lane
  const int g    = lane >> 4;       // lane-half
  const int wid  = tid >> 5;        // wave id = which head of the KV group

  const int bid = blockIdx.x;
  const int qt2 = bid % qtiles2;                  // 32-query tile index
  const int kh  = (bid / qtiles2) % NKH;          // KV head
  const int b   = bid / (qtiles2 * NKH);
  const int h   = kh * HPK + wid;                 // this wave's query head

  const int seq0 = cu[b];
  int qtok[2];
  qtok[0] = seq0 + qt2 * 32 + n;
  qtok[1] = qtok[0] + 16;

  // ---- Q^T as WMMA B-fragments for both 16-query sets (scale*log2e folded) ----
  // B layout: lane = col n; halves 0..7 -> d = dc*32 + 8g + 0..7, 8..15 -> dc*32 + 16 + 8g + 0..7.
  H16 Qf[2][4];
#pragma unroll
  for (int j = 0; j < 2; ++j) {
    const float* qrow = q + (size_t)qtok[j] * QSTR + h * HD;
#pragma unroll
    for (int dc = 0; dc < 4; ++dc) {
      const float* p = qrow + dc * 32 + g * 8;
#pragma unroll
      for (int t = 0; t < 4; ++t) {
        float4 x = ld_f4(p + 4 * t + (t >> 1) * 8);
        Qf[j][dc].h[4 * t + 0] = (_Float16)(x.x * QK_SCALE_LOG2);
        Qf[j][dc].h[4 * t + 1] = (_Float16)(x.y * QK_SCALE_LOG2);
        Qf[j][dc].h[4 * t + 2] = (_Float16)(x.z * QK_SCALE_LOG2);
        Qf[j][dc].h[4 * t + 3] = (_Float16)(x.w * QK_SCALE_LOG2);
      }
    }
  }

  // O^T accumulators: per query set, 8 d-blocks (lane = query col, vgpr = d row)
  F8 O[2][8];
#pragma unroll
  for (int j = 0; j < 2; ++j)
#pragma unroll
    for (int db = 0; db < 8; ++db)
#pragma unroll
      for (int r = 0; r < 8; ++r) O[j][db].f[r] = 0.0f;

  float m_run[2] = {NEG_BIG, NEG_BIG};
  float l_run[2] = {0.0f, 0.0f};

  // Cooperative fill assignments (whole 128-thread block):
  const int fk_key = tid >> 2;        // K: thread -> (key row, 32-d segment)
  const int fk_seg = tid & 3;
  // V: thread -> one d column (loads 32 keys, writes one transposed LDS row)

  const v8f zero8 = {0.f, 0.f, 0.f, 0.f, 0.f, 0.f, 0.f, 0.f};
  const int nkt = qt2 + 1;   // uniform across the whole block

  for (int kt = 0; kt < nkt; ++kt) {
    const int kb = seq0 + kt * 32;

    // ---- issue global loads for this tile before the barrier (overlap) ----
    const float* kp = k + (size_t)(kb + fk_key) * KVSTR + kh * HD + fk_seg * 32;
    float4 kx[8];
#pragma unroll
    for (int t = 0; t < 8; ++t) kx[t] = ld_f4(kp + 4 * t);

    const float* vp = v + (size_t)kb * KVSTR + kh * HD + tid;
    float vx[32];
#pragma unroll
    for (int i = 0; i < 32; ++i) vx[i] = vp[(size_t)i * KVSTR];

    if (kt + 1 < nkt) {   // uniform: speculative prefetch of next K/V tile
      __builtin_prefetch(kp + 32 * KVSTR, 0, 0);
      __builtin_prefetch(vp + 32 * KVSTR, 0, 0);
    }

    __syncthreads();   // previous tile's LDS reads complete

    // K tile -> LDS [key][d], f16
    {
      unsigned kw[16];
#pragma unroll
      for (int t = 0; t < 8; ++t) {
        kw[2 * t]     = pack_h2(kx[t].x, kx[t].y);
        kw[2 * t + 1] = pack_h2(kx[t].z, kx[t].w);
      }
      uint4* kdst = (uint4*)&ldsK[fk_key * KSTR + fk_seg * 32];
      kdst[0] = make_uint4(kw[0],  kw[1],  kw[2],  kw[3]);
      kdst[1] = make_uint4(kw[4],  kw[5],  kw[6],  kw[7]);
      kdst[2] = make_uint4(kw[8],  kw[9],  kw[10], kw[11]);
      kdst[3] = make_uint4(kw[12], kw[13], kw[14], kw[15]);
    }
    // V tile -> LDS transposed [d][key], f16
    {
      unsigned vw[16];
#pragma unroll
      for (int i = 0; i < 16; ++i) vw[i] = pack_h2(vx[2 * i], vx[2 * i + 1]);
      uint4* vdst = (uint4*)&ldsV[tid * VSTR];
      vdst[0] = make_uint4(vw[0],  vw[1],  vw[2],  vw[3]);
      vdst[1] = make_uint4(vw[4],  vw[5],  vw[6],  vw[7]);
      vdst[2] = make_uint4(vw[8],  vw[9],  vw[10], vw[11]);
      vdst[3] = make_uint4(vw[12], vw[13], vw[14], vw[15]);
    }

    __syncthreads();   // tile visible to all waves

    // ---- S^T: 2 key blocks x 2 query sets, accumulated over 4 D-chunks ----
    v8f st[2][2];  // [keyblock][qset]
    st[0][0] = zero8; st[0][1] = zero8; st[1][0] = zero8; st[1][1] = zero8;
#pragma unroll
    for (int dc = 0; dc < 4; ++dc) {
      H16 ka0, ka1;   // A layout: lane row = key; halves -> d 8g+0..7 / 16+8g+0..7
      lds_frag(ka0, &ldsK[(n)      * KSTR + dc * 32 + g * 8]);
      lds_frag(ka1, &ldsK[(n + 16) * KSTR + dc * 32 + g * 8]);
      st[0][0] = __builtin_amdgcn_wmma_f32_16x16x32_f16(false, ka0.v, false, Qf[0][dc].v,
                                                        (short)0, st[0][0], false, false);
      st[0][1] = __builtin_amdgcn_wmma_f32_16x16x32_f16(false, ka0.v, false, Qf[1][dc].v,
                                                        (short)0, st[0][1], false, false);
      st[1][0] = __builtin_amdgcn_wmma_f32_16x16x32_f16(false, ka1.v, false, Qf[0][dc].v,
                                                        (short)0, st[1][0], false, false);
      st[1][1] = __builtin_amdgcn_wmma_f32_16x16x32_f16(false, ka1.v, false, Qf[1][dc].v,
                                                        (short)0, st[1][1], false, false);
    }

    F8 s[2][2];
#pragma unroll
    for (int kb2 = 0; kb2 < 2; ++kb2)
#pragma unroll
      for (int j = 0; j < 2; ++j) s[kb2][j].v = st[kb2][j];

    // Causal mask: only the diagonal (last) tile; per-lane cndmask, uniform branch
    if (kt == nkt - 1) {
#pragma unroll
      for (int j = 0; j < 2; ++j)
#pragma unroll
        for (int r = 0; r < 8; ++r) {
          const int key0 = kb + g * 8 + r;     // S^T row -> key token
          s[0][j].f[r] = (key0      > qtok[j]) ? NEG_BIG : s[0][j].f[r];
          s[1][j].f[r] = (key0 + 16 > qtok[j]) ? NEG_BIG : s[1][j].f[r];
        }
    }

    // ---- online softmax per query set + P^T B-fragments ----
    H16 Pb[2];
#pragma unroll
    for (int j = 0; j < 2; ++j) {
      float tm = NEG_BIG;
#pragma unroll
      for (int r = 0; r < 8; ++r) tm = fmaxf(tm, fmaxf(s[0][j].f[r], s[1][j].f[r]));
      tm = fmaxf(tm, __shfl_xor(tm, 16, 32));
      const float mn = fmaxf(m_run[j], tm);
      const float al = __builtin_amdgcn_exp2f(m_run[j] - mn);

      float p0[8], p1[8];
      float rs = 0.0f;
#pragma unroll
      for (int r = 0; r < 8; ++r) {
        p0[r] = __builtin_amdgcn_exp2f(s[0][j].f[r] - mn);
        p1[r] = __builtin_amdgcn_exp2f(s[1][j].f[r] - mn);
        rs += p0[r] + p1[r];
      }
      rs += __shfl_xor(rs, 16, 32);
      l_run[j] = l_run[j] * al + rs;
      m_run[j] = mn;

#pragma unroll
      for (int db = 0; db < 8; ++db)
#pragma unroll
        for (int r = 0; r < 8; ++r) O[j][db].f[r] *= al;

      // pack f16 -> B layout (halves = keys 8g+0..7 / 16+8g+0..7)
      unsigned pk0[4], pk1[4];
#pragma unroll
      for (int t = 0; t < 4; ++t) {
        pk0[t] = pack_h2(p0[2 * t], p0[2 * t + 1]);   // keys 8g+0..7
        pk1[t] = pack_h2(p1[2 * t], p1[2 * t + 1]);   // keys 16+8g+0..7
      }
#pragma unroll
      for (int t = 0; t < 4; ++t) {
        const unsigned o0 = (unsigned)__shfl_xor((int)pk0[t], 16, 32);
        const unsigned o1 = (unsigned)__shfl_xor((int)pk1[t], 16, 32);
        Pb[j].u[t]     = pk0[t];
        Pb[j].u[4 + t] = pk1[t];
      }
    }

    // ---- O^T += V^T x P^T : each V^T fragment (from LDS) feeds both query sets ----
#pragma unroll
    for (int db = 0; db < 8; ++db) {
      H16 vf;   // A layout: lane row d = db*16+n; halves -> keys 8g+i / 16+8g+i
      lds_frag(vf, &ldsV[(db * 16 + n) * VSTR + g * 8]);
      O[0][db].v = __builtin_amdgcn_wmma_f32_16x16x32_f16(false, vf.v, false, Pb[0].v,
                                                          (short)0, O[0][db].v, false, false);
      O[1][db].v = __builtin_amdgcn_wmma_f32_16x16x32_f16(false, vf.v, false, Pb[1].v,
                                                          (short)0, O[1][db].v, false, false);
    }
  }

  // ---- normalize and store both query sets: rows are d = db*16 + 8g + r ----
#pragma unroll
  for (int j = 0; j < 2; ++j) {
    const float inv = 1.0f / l_run[j];
    float* orow = out + (size_t)qtok[j] * QSTR + h * HD + g * 8;
#pragma unroll
    for (int db = 0; db < 8; ++db) {
      float4 lo, hi;
      lo.x = O[j][db].f[0] * inv; lo.y = O[j][db].f[1] * inv;
      lo.z = O[j][db].f[2] * inv; lo.w = O[j][db].f[3] * inv;
      hi.x = O[j][db].f[4] * inv; hi.y = O[j][db].f[5] * inv;
      hi.z = O[j][db].f[6] * inv; hi.w = O[j][db].f[7] * inv;
      *(float4*)(orow + db * 16)     = lo;
      *(float4*)(orow + db * 16 + 4) = hi;
    }
  }
}

extern "C" void kernel_launch(void* const* d_in, const int* in_sizes, int n_in,
                              void* d_out, int out_size, void* d_ws, size_t ws_size,
                              hipStream_t stream) {
  (void)n_in; (void)out_size; (void)d_ws; (void)ws_size;
  const float* q  = (const float*)d_in[0];
  const float* k  = (const float*)d_in[1];
  const float* v  = (const float*)d_in[2];
  const int*   cu = (const int*)d_in[3];
  float* out = (float*)d_out;

  const int T = in_sizes[0] / (NH * HD);   // total tokens
  const int B = in_sizes[3] - 1;
  const int S = T / B;                     // equal-length segments
  const int qtiles2 = S / 32;

  // 1 block = 1 GQA group (4 heads) x one 32-query tile
  const int blocks = B * NKH * qtiles2;
  dim3 grid(blocks), block(128);
  fa_fwd_causal_gqa<<<grid, block, 0, stream>>>(q, k, v, cu, out, qtiles2);
}
